// MultiHeadAttention_4844723110310
// MI455X (gfx1250) — matrix operand order repaired and checked by reading, not compiled, not run on hardware
//
#include <hip/hip_runtime.h>
#include <hip/hip_bf16.h>

// ---------------- problem constants ----------------
static constexpr int Bn   = 2;
static constexpr int Tn   = 2048;
static constexpr int Cn   = 1024;
static constexpr int Hn   = 16;
static constexpr int HDn  = 64;
static constexpr int HD2n = 32;
static constexpr int Mtot = Bn * Tn;     // 4096
static constexpr int N1   = 3 * Cn;      // 3072

// ---------------- types ----------------
typedef __attribute__((ext_vector_type(16))) __bf16        v16bf;
typedef __attribute__((ext_vector_type(8)))  float         v8f;
typedef __attribute__((ext_vector_type(4)))  unsigned int  u32x4;

union FragU { v16bf v; u32x4 q[2]; };
__device__ __forceinline__ v16bf make_frag(u32x4 a, u32x4 b) {
  FragU u; u.q[0] = a; u.q[1] = b; return u.v;
}

struct F2 { float x, y; };

__device__ __forceinline__ unsigned short f2bf(float f) {
  unsigned int u = __float_as_uint(f);
  u = (u + 0x7FFFu + ((u >> 16) & 1u)) >> 16;
  return (unsigned short)u;
}

#if defined(__has_builtin) && __has_builtin(__builtin_amdgcn_cvt_pk_bf16_f32)
typedef __attribute__((ext_vector_type(2))) __bf16 v2bf;
__device__ __forceinline__ unsigned int pack_bf16x2(float lo, float hi) {
  v2bf r = __builtin_amdgcn_cvt_pk_bf16_f32(lo, hi);
  return __builtin_bit_cast(unsigned int, r);
}
#else
__device__ __forceinline__ unsigned int pack_bf16x2(float lo, float hi) {
  return (unsigned int)f2bf(lo) | ((unsigned int)f2bf(hi) << 16);
}
#endif

#define WMMA_BF16(A, B, C) \
  __builtin_amdgcn_wmma_f32_16x16x32_bf16(false, (A), false, (B), (short)0, (C), false, false)

// ---------------- async global->LDS (CDNA5) ----------------
#if defined(__has_builtin) && __has_builtin(__builtin_amdgcn_global_load_async_to_lds_b128)
#define HAS_ASYNC 1
#else
#define HAS_ASYNC 0
#endif

#if HAS_ASYNC
// From the compiler diagnostic: param 1 is AS1 (printed "__device__") pointer to a
// 16-byte int vector. Param 2 is the LDS (AS3) side, same element type.
typedef int v4i __attribute__((vector_size(16)));
typedef __attribute__((address_space(1))) v4i g_v4i;
typedef __attribute__((address_space(3))) v4i l_v4i;
__device__ __forceinline__ void async_cp16(void* lds, const void* g) {
  // go through integers: low 32 bits of a generic LDS pointer are the LDS offset
  g_v4i* gp = (g_v4i*)(unsigned long long)g;
  l_v4i* lp = (l_v4i*)(unsigned int)(unsigned long long)lds;
  __builtin_amdgcn_global_load_async_to_lds_b128(gp, lp, 0, 0);
}
#if __has_builtin(__builtin_amdgcn_s_wait_asynccnt)
#define WAIT_ASYNC(n) __builtin_amdgcn_s_wait_asynccnt(n)
#else
#define WAIT_ASYNC(n) asm volatile("s_wait_asynccnt %0" ::"n"(n))
#endif
#endif

// =====================================================================
// Stage 1: qkv = x @ Wqkv + b, RoPE on q,k; write bf16 Qr/Kr [B][H][T][HD]
//          and V transposed Vt [B][H][HD][T].
// Block: 256 thr (8 waves). WG tile 64(M) x 128(N); wave tile 16 x 64.
// =====================================================================
__global__ __launch_bounds__(256) void qkv_rope_kernel(
    const float* __restrict__ x, const float* __restrict__ fcos,
    const float* __restrict__ fsin, const float* __restrict__ Wqkv,
    const float* __restrict__ bqkv,
    unsigned short* __restrict__ Qr, unsigned short* __restrict__ Kr,
    unsigned short* __restrict__ Vt)
{
  __shared__ __align__(16) unsigned short At[64 * 40];    // A tile bf16 [row][k], stride 40
  __shared__ __align__(16) unsigned short Wt[128 * 40];   // W tile bf16 transposed [n][k]

  const int tid  = threadIdx.x;
  const int w    = tid >> 5;
  const int lane = tid & 31;
  const int rsel = lane & 15;
  const int hi   = lane >> 4;
  const int mbase = blockIdx.y * 64;
  const int nbase = blockIdx.x * 128;
  const int mw = (w & 3) * 16;        // wave row offset in tile
  const int nw = (w >> 2) * 64;       // wave col offset in tile

  v8f acc[4] = {};

  for (int k0 = 0; k0 < Cn; k0 += 32) {
    // ---- stage A tile (64x32 f32 -> bf16) ----
#pragma unroll
    for (int i = 0; i < 4; ++i) {
      int p = tid + 256 * i;                // 0..1023 packs
      int row = p >> 4, kp = p & 15;
      F2 f = *reinterpret_cast<const F2*>(&x[(size_t)(mbase + row) * Cn + k0 + 2 * kp]);
      *reinterpret_cast<unsigned int*>(&At[row * 40 + 2 * kp]) = pack_bf16x2(f.x, f.y);
    }
    // ---- stage W tile (32x128 f32 -> bf16, transposed) ----
#pragma unroll
    for (int i = 0; i < 8; ++i) {
      int p = tid + 256 * i;                // 0..2047 packs
      int n = p & 127, kp = p >> 7;         // kp 0..15
      float a = Wqkv[(size_t)(k0 + 2 * kp) * N1 + nbase + n];
      float b = Wqkv[(size_t)(k0 + 2 * kp + 1) * N1 + nbase + n];
      *reinterpret_cast<unsigned int*>(&Wt[n * 40 + 2 * kp]) = pack_bf16x2(a, b);
    }
    __syncthreads();

    const u32x4* arow = reinterpret_cast<const u32x4*>(&At[(mw + rsel) * 40]);
    v16bf af = make_frag(arow[2 * hi], arow[2 * hi + 1]);   // K contiguous per half-wave
#pragma unroll
    for (int j = 0; j < 4; ++j) {
      const u32x4* brow = reinterpret_cast<const u32x4*>(&Wt[(nw + j * 16 + rsel) * 40]);
      v16bf bf = make_frag(brow[2 * hi], brow[2 * hi + 1]);  // K contiguous per half-wave
      acc[j] = WMMA_BF16(af, bf, acc[j]);
    }
    __syncthreads();
  }

  // ---- epilogue: bias + RoPE + scatter ----
  const int ncol0 = nbase + nw;             // 64-aligned -> one (section, head)
  const int sect  = ncol0 >> 10;            // 0=q 1=k 2=v
  const int hh    = (ncol0 & 1023) >> 6;

  if (sect < 2) {
    unsigned short* dst = (sect == 0) ? Qr : Kr;
#pragma unroll
    for (int r = 0; r < 8; ++r) {
      int m = mbase + mw + r + 8 * hi;
      int b = m >> 11, t = m & (Tn - 1);
#pragma unroll
      for (int j = 0; j < 2; ++j) {
        int hd = j * 16 + rsel;                       // 0..31
        float u1 = acc[j][r]     + bqkv[ncol0 + j * 16 + rsel];
        float u2 = acc[j + 2][r] + bqkv[ncol0 + (j + 2) * 16 + rsel];
        float c = fcos[t * HD2n + hd];
        float s = fsin[t * HD2n + hd];
        size_t base = ((size_t)(b * Hn + hh) * Tn + t) * HDn;
        dst[base + hd]      = f2bf(u1 * c - u2 * s);
        dst[base + hd + 32] = f2bf(u1 * s + u2 * c);
      }
    }
  } else {
#pragma unroll
    for (int r = 0; r < 8; ++r) {
      int m = mbase + mw + r + 8 * hi;
      int b = m >> 11, t = m & (Tn - 1);
#pragma unroll
      for (int j = 0; j < 4; ++j) {
        int hd = j * 16 + rsel;
        float v = acc[j][r] + bqkv[ncol0 + j * 16 + rsel];
        Vt[((size_t)(b * Hn + hh) * HDn + hd) * Tn + t] = f2bf(v);
      }
    }
  }
}

// =====================================================================
// Stage 2: causal flash attention per (b,h). Block 128 thr (4 waves),
// 64 query rows per block (16 per wave), KV tiles of 64 staged in LDS.
// Double-buffered with GLOBAL_LOAD_ASYNC_TO_LDS_B128 when available.
// =====================================================================
#if HAS_ASYNC
static constexpr int NBUF = 2;
#else
static constexpr int NBUF = 1;
#endif

__global__ __launch_bounds__(128) void attn_kernel(
    const unsigned short* __restrict__ Qr, const unsigned short* __restrict__ Kr,
    const unsigned short* __restrict__ Vt, unsigned short* __restrict__ Y)
{
  __shared__ __align__(16) unsigned short Kt[NBUF][64 * 72];   // [key][hd]
  __shared__ __align__(16) unsigned short Vl[NBUF][64 * 72];   // [hd][key]
  __shared__ __align__(16) unsigned short Pt[4][16 * 72];      // per-wave P [row][key]

  const int tid  = threadIdx.x;
  const int w    = tid >> 5;
  const int lane = tid & 31;
  const int rsel = lane & 15;
  const int hi   = lane >> 4;
  const int bh   = blockIdx.y;          // b*H + h
  const int qt   = blockIdx.x;          // 64-row query tile
  const int qw   = qt * 64 + w * 16;    // wave query base (t units)
  const float scale = 0.125f;           // 1/sqrt(64)

  // preload Q fragments (kept in registers for whole kv loop)
  v16bf qf[2];
  {
    const u32x4* qp = reinterpret_cast<const u32x4*>(
        Qr + ((size_t)bh * Tn + qw + rsel) * HDn);
#pragma unroll
    for (int c = 0; c < 2; ++c)
      qf[c] = make_frag(qp[c * 4 + 2 * hi], qp[c * 4 + 2 * hi + 1]);
  }

  v8f o[4] = {};
  float mrow[8], lrow[8];
#pragma unroll
  for (int r = 0; r < 8; ++r) { mrow[r] = -__builtin_inff(); lrow[r] = 0.0f; }

#if HAS_ASYNC
  // issue one KV tile: 8 async b128 per thread (K: 4, V: 4)
  auto stage_async = [&](int jt, int buf) {
    const int kv = jt * 64;
#pragma unroll
    for (int i = 0; i < 4; ++i) {
      int p = tid + 128 * i;            // 0..511
      int row = p >> 3, seg = p & 7;    // 64 rows x 8 x 16B segs
      async_cp16(&Kt[buf][row * 72 + seg * 8],
                 Kr + (size_t)(bh * Tn + kv + row) * HDn + seg * 8);
      async_cp16(&Vl[buf][row * 72 + seg * 8],
                 Vt + ((size_t)bh * HDn + row) * Tn + kv + seg * 8);
    }
  };
  stage_async(0, 0);
#endif

  for (int jt = 0; jt <= qt; ++jt) {
    const int kv = jt * 64;
#if HAS_ASYNC
    const int buf = jt & 1;
    if (jt < qt) {                 // prefetch next tile into other buffer
      stage_async(jt + 1, buf ^ 1);
      WAIT_ASYNC(8);               // oldest tile (jt) complete; next may be in flight
    } else {
      WAIT_ASYNC(0);
    }
    __syncthreads();
#else
    const int buf = 0;
#pragma unroll
    for (int i = 0; i < 4; ++i) {
      int p = tid + 128 * i;
      int row = p >> 3, seg = p & 7;
      u32x4 kq = reinterpret_cast<const u32x4*>(Kr)[(size_t)(bh * Tn + kv + row) * 8 + seg];
      *reinterpret_cast<u32x4*>(&Kt[0][row * 72 + seg * 8]) = kq;
      u32x4 vq = reinterpret_cast<const u32x4*>(Vt)[((size_t)bh * HDn + row) * (Tn / 8) + (kv >> 3) + seg];
      *reinterpret_cast<u32x4*>(&Vl[0][row * 72 + seg * 8]) = vq;
    }
    __syncthreads();
#endif

    // ---- S = Q K^T (wave: 16 x 64) ----
    v8f sa[4] = {};
#pragma unroll
    for (int c = 0; c < 2; ++c) {
#pragma unroll
      for (int n = 0; n < 4; ++n) {
        const u32x4* krow = reinterpret_cast<const u32x4*>(&Kt[buf][(n * 16 + rsel) * 72]);
        v16bf kf = make_frag(krow[c * 4 + 2 * hi], krow[c * 4 + 2 * hi + 1]);
        sa[n] = WMMA_BF16(qf[c], kf, sa[n]);
      }
    }

    // ---- online softmax in C-layout (row = acc index, col = lane) ----
#pragma unroll
    for (int r = 0; r < 8; ++r) {
      const int qg = qw + r + 8 * hi;
      float sv[4];
      float vmax = -__builtin_inff();
#pragma unroll
      for (int n = 0; n < 4; ++n) {
        int kg = kv + n * 16 + rsel;
        float s = sa[n][r] * scale;
        if (kg > qg) s = -__builtin_inff();
        sv[n] = s;
        vmax = fmaxf(vmax, s);
      }
#pragma unroll
      for (int off = 8; off >= 1; off >>= 1)
        vmax = fmaxf(vmax, __shfl_xor(vmax, off, 32));
      float mnew = fmaxf(mrow[r], vmax);
      float rs = 0.0f;
#pragma unroll
      for (int n = 0; n < 4; ++n) {
        float p = __expf(sv[n] - mnew);
        rs += p;
        Pt[w][(r + 8 * hi) * 72 + n * 16 + rsel] = f2bf(p);
      }
#pragma unroll
      for (int off = 8; off >= 1; off >>= 1)
        rs += __shfl_xor(rs, off, 32);
      float alpha = __expf(mrow[r] - mnew);
      mrow[r] = mnew;
      lrow[r] = lrow[r] * alpha + rs;
#pragma unroll
      for (int nh = 0; nh < 4; ++nh) o[nh][r] *= alpha;
    }

    // ---- O += P V (P re-read from LDS in A-fragment layout) ----
#pragma unroll
    for (int c = 0; c < 2; ++c) {
      const u32x4* prow = reinterpret_cast<const u32x4*>(&Pt[w][rsel * 72]);
      v16bf pf = make_frag(prow[c * 4 + 2 * hi], prow[c * 4 + 2 * hi + 1]);
#pragma unroll
      for (int nh = 0; nh < 4; ++nh) {
        const u32x4* vrow = reinterpret_cast<const u32x4*>(&Vl[buf][(nh * 16 + rsel) * 72]);
        v16bf vf = make_frag(vrow[c * 4 + 2 * hi], vrow[c * 4 + 2 * hi + 1]);
        o[nh] = WMMA_BF16(pf, vf, o[nh]);
      }
    }
    __syncthreads();   // all waves done reading this buffer
  }

  // ---- normalize & store Y bf16 [B][T][C] ----
  const int b = bh >> 4, h = bh & 15;
#pragma unroll
  for (int r = 0; r < 8; ++r) {
    int t = qw + r + 8 * hi;
    float inv = 1.0f / lrow[r];
#pragma unroll
    for (int nh = 0; nh < 4; ++nh) {
      float val = o[nh][r] * inv;
      Y[(size_t)(b * Tn + t) * Cn + h * HDn + nh * 16 + rsel] = f2bf(val);
    }
  }
}

// =====================================================================
// Stage 3: out = Y @ Wproj + bproj (f32 out). Same tiling as stage 1.
// =====================================================================
__global__ __launch_bounds__(256) void proj_kernel(
    const unsigned short* __restrict__ Y, const float* __restrict__ Wp,
    const float* __restrict__ bp, float* __restrict__ out)
{
  __shared__ __align__(16) unsigned short At[64 * 40];
  __shared__ __align__(16) unsigned short Wt[128 * 40];

  const int tid  = threadIdx.x;
  const int w    = tid >> 5;
  const int lane = tid & 31;
  const int rsel = lane & 15;
  const int hi   = lane >> 4;
  const int mbase = blockIdx.y * 64;
  const int nbase = blockIdx.x * 128;
  const int mw = (w & 3) * 16;
  const int nw = (w >> 2) * 64;

  v8f acc[4] = {};

  for (int k0 = 0; k0 < Cn; k0 += 32) {
#pragma unroll
    for (int i = 0; i < 4; ++i) {
      int p = tid + 256 * i;
      int row = p >> 4, kp = p & 15;
      *reinterpret_cast<unsigned int*>(&At[row * 40 + 2 * kp]) =
          *reinterpret_cast<const unsigned int*>(&Y[(size_t)(mbase + row) * Cn + k0 + 2 * kp]);
    }
#pragma unroll
    for (int i = 0; i < 8; ++i) {
      int p = tid + 256 * i;
      int n = p & 127, kp = p >> 7;
      float a = Wp[(size_t)(k0 + 2 * kp) * Cn + nbase + n];
      float b = Wp[(size_t)(k0 + 2 * kp + 1) * Cn + nbase + n];
      *reinterpret_cast<unsigned int*>(&Wt[n * 40 + 2 * kp]) = pack_bf16x2(a, b);
    }
    __syncthreads();

    const u32x4* arow = reinterpret_cast<const u32x4*>(&At[(mw + rsel) * 40]);
    v16bf af = make_frag(arow[2 * hi], arow[2 * hi + 1]);
#pragma unroll
    for (int j = 0; j < 4; ++j) {
      const u32x4* brow = reinterpret_cast<const u32x4*>(&Wt[(nw + j * 16 + rsel) * 40]);
      v16bf bf = make_frag(brow[2 * hi], brow[2 * hi + 1]);
      acc[j] = WMMA_BF16(af, bf, acc[j]);
    }
    __syncthreads();
  }

#pragma unroll
  for (int r = 0; r < 8; ++r) {
    int m = mbase + mw + r + 8 * hi;
#pragma unroll
    for (int j = 0; j < 4; ++j) {
      int n = nbase + nw + j * 16 + rsel;
      out[(size_t)m * Cn + n] = acc[j][r] + bp[n];
    }
  }
}

// =====================================================================
extern "C" void kernel_launch(void* const* d_in, const int* in_sizes, int n_in,
                              void* d_out, int out_size, void* d_ws, size_t ws_size,
                              hipStream_t stream) {
  const float* x    = (const float*)d_in[0];
  const float* fc   = (const float*)d_in[1];
  const float* fs   = (const float*)d_in[2];
  const float* Wqkv = (const float*)d_in[3];
  const float* bqkv = (const float*)d_in[4];
  const float* Wp   = (const float*)d_in[5];
  const float* bp   = (const float*)d_in[6];
  float* out = (float*)d_out;

  const size_t elems = (size_t)Bn * Hn * Tn * HDn;   // 4M bf16 elements each
  unsigned short* Qr = (unsigned short*)d_ws;
  unsigned short* Kr = Qr + elems;
  unsigned short* Vt = Kr + elems;
  unsigned short* Y  = Vt + elems;                   // total 32 MB of ws

  dim3 g1(N1 / 128, Mtot / 64);    // (24, 64)
  qkv_rope_kernel<<<g1, dim3(256), 0, stream>>>(x, fc, fs, Wqkv, bqkv, Qr, Kr, Vt);

  dim3 g2(Tn / 64, Bn * Hn);       // (32, 32)
  attn_kernel<<<g2, dim3(128), 0, stream>>>(Qr, Kr, Vt, Y);

  dim3 g3(Cn / 128, Mtot / 64);    // (8, 64)
  proj_kernel<<<g3, dim3(256), 0, stream>>>(Y, Wp, bp, out);
}
